// GCN_5325759447714
// MI455X (gfx1250) — matrix operand order repaired and checked by reading, compiled, not run on hardware
//
#include <hip/hip_runtime.h>

#define NODES 50000
#define EDGES 800000

typedef __attribute__((ext_vector_type(16))) __bf16    v16bf;
typedef __attribute__((ext_vector_type(8)))  float     v8f;
typedef __attribute__((ext_vector_type(8)))  unsigned  v8u;
typedef __attribute__((ext_vector_type(4)))  unsigned  v4u;

// ---------------- degree / normalization ----------------
__global__ __launch_bounds__(256) void k_deg_init(float* __restrict__ deg, int n) {
    int i = blockIdx.x * blockDim.x + threadIdx.x;
    if (i < n) deg[i] = 1.0f;              // self-loop contributes 1 to every node
}

__global__ __launch_bounds__(256) void k_deg_count(const int* __restrict__ dst,
                                                   float* __restrict__ deg, int e) {
    int i = blockIdx.x * blockDim.x + threadIdx.x;
    if (i < e) atomicAdd(&deg[dst[i]], 1.0f);
}

__global__ __launch_bounds__(256) void k_deg_rsqrt(float* __restrict__ deg, int n) {
    int i = blockIdx.x * blockDim.x + threadIdx.x;
    if (i < n) deg[i] = rsqrtf(deg[i]);    // deg >= 1 always (self loops)
}

// ---------------- GEMM: Y[n x DOUT] = X[n x 128] @ X W[128 x DOUT] (+bias) ----------------
// 256 threads = 8 wave32; block tile = 128 rows x DOUT cols; bf16 WMMA, fp32 accum.
template <int DOUT, bool ADD_BIAS>
__global__ __launch_bounds__(256) void k_gemm_bf16(const float* __restrict__ X,
                                                   const float* __restrict__ W,
                                                   const float* __restrict__ bias,
                                                   float* __restrict__ Y, int n) {
    constexpr int NT = DOUT / 16;
    // A tile: 128 rows x 128 K bf16, row padded to 136 (rows stay 16B aligned, banks spread)
    __shared__ __bf16 As[128][136];
    // B fragments pre-swizzled: Bs[ks][half][n][v] = bf16 pair (W[2*kp][n], W[2*kp+1][n]),
    // kp = ks*16 + (v&4)*2 + half*4 + (v&3).  Stride 12 dwords: 48B -> 16B-aligned b128 loads, gcd(12,64)=4
    // so the 16 columns of a half-wave land on distinct banks. Fragment = one 32B read.
    __shared__ unsigned Bs[4][2][DOUT][12];

    const int tid  = threadIdx.x;
    const int row0 = blockIdx.x * 128;

    // W (fp32, row-major K x DOUT) -> swizzled LDS bf16 pairs; n fastest -> coalesced
    for (int i = tid; i < 4 * 2 * 8 * DOUT; i += 256) {
        int nn   = i % DOUT;
        int r    = i / DOUT;
        int v    = r & 7;
        int half = (r >> 3) & 1;
        int ks   = r >> 4;
        int kp   = ks * 16 + (v & 4) * 2 + half * 4 + (v & 3);
        union { unsigned u; __bf16 h[2]; } p;
        p.h[0] = (__bf16)W[(2 * kp + 0) * DOUT + nn];
        p.h[1] = (__bf16)W[(2 * kp + 1) * DOUT + nn];
        Bs[ks][half][nn][v] = p.u;
    }
    // X tile (fp32) -> LDS bf16, float4 coalesced loads
    for (int i = tid; i < 128 * 32; i += 256) {
        int r = i >> 5, c4 = i & 31;
        int gr = row0 + r;
        float4 v = make_float4(0.f, 0.f, 0.f, 0.f);
        if (gr < n) v = reinterpret_cast<const float4*>(X + (size_t)gr * 128)[c4];
        __bf16* d = &As[r][c4 * 4];
        d[0] = (__bf16)v.x; d[1] = (__bf16)v.y; d[2] = (__bf16)v.z; d[3] = (__bf16)v.w;
    }
    __syncthreads();

    const int wave = tid >> 5, lane = tid & 31;
    const int half = lane >> 4, nl = lane & 15;
    const int mr   = wave * 16 + nl;     // A row for this lane

    // A fragments (16x32 bf16): lanes 0-15 hold K blocks [0..7],[16..23]; lanes 16-31 the others
    v16bf afrag[4];
#pragma unroll
    for (int ks = 0; ks < 4; ++ks) {
        const __bf16* arow = &As[mr][0];
        v4u lo = *reinterpret_cast<const v4u*>(arow + ks * 32 + half * 8);
        v4u hi = *reinterpret_cast<const v4u*>(arow + ks * 32 + 16 + half * 8);
        v8u au;
#pragma unroll
        for (int i = 0; i < 4; ++i) { au[i] = lo[i]; au[4 + i] = hi[i]; }
        afrag[ks] = __builtin_bit_cast(v16bf, au);
    }

    v8f acc[NT] = {};
#pragma unroll
    for (int t = 0; t < NT; ++t) {
        const int nn = t * 16 + nl;      // B column for this lane
#pragma unroll
        for (int ks = 0; ks < 4; ++ks) {
            v8u bu = *reinterpret_cast<const v8u*>(&Bs[ks][half][nn][0]);
            v16bf bfrag = __builtin_bit_cast(v16bf, bu);
            acc[t] = __builtin_amdgcn_wmma_f32_16x16x32_bf16(
                false, afrag[ks], false, bfrag, (short)0, acc[t], false, false);
        }
    }

    // C/D layout: VGPR r holds M = half*8 + r, N = lane&15
#pragma unroll
    for (int t = 0; t < NT; ++t) {
        const int nOut = t * 16 + nl;
        const float badd = ADD_BIAS ? bias[nOut] : 0.0f;
#pragma unroll
        for (int r = 0; r < 8; ++r) {
            int gm = row0 + wave * 16 + half * 8 + r;
            if (gm < n) Y[(size_t)gm * DOUT + nOut] = acc[t][r] + badd;
        }
    }
}

// ---------------- aggregation ----------------
// O[i] = H[i] * dinv[i]^2   (self-loop term; also initializes O)
__global__ __launch_bounds__(256) void k_self_init(const float* __restrict__ H,
                                                   const float* __restrict__ dinv,
                                                   float* __restrict__ O, int n) {
    int idx = blockIdx.x * blockDim.x + threadIdx.x;
    int node = idx >> 5, c = idx & 31;
    if (node >= n) return;
    float s = dinv[node]; s = s * s;
    float4 v = reinterpret_cast<const float4*>(H + (size_t)node * 128)[c];
    float4 r = make_float4(v.x * s, v.y * s, v.z * s, v.w * s);
    reinterpret_cast<float4*>(O + (size_t)node * 128)[c] = r;
}

// one wave32 per edge: 32 lanes x float4 = full 128-float row; f32 atomics resolve in L2.
// edge index is forced wave-uniform via readfirstlane -> src/dst/dinv become scalar loads.
__global__ __launch_bounds__(256) void k_edge_scatter(const int* __restrict__ src,
                                                      const int* __restrict__ dst,
                                                      const float* __restrict__ dinv,
                                                      const float* __restrict__ H,
                                                      float* __restrict__ O, int e) {
    int lane = threadIdx.x & 31;
    int edge = __builtin_amdgcn_readfirstlane(
        (int)((blockIdx.x * blockDim.x + threadIdx.x) >> 5));
    if (edge >= e) return;
    int s = src[edge];                 // uniform -> scalar load
    int d = dst[edge];
    float nrm = dinv[s] * dinv[d];
    float4 v = reinterpret_cast<const float4*>(H + (size_t)s * 128)[lane];
    float* o = O + (size_t)d * 128 + lane * 4;
    atomicAdd(o + 0, v.x * nrm);
    atomicAdd(o + 1, v.y * nrm);
    atomicAdd(o + 2, v.z * nrm);
    atomicAdd(o + 3, v.w * nrm);
}

__global__ __launch_bounds__(256) void k_bias_relu(float* __restrict__ O,
                                                   const float* __restrict__ b, int n) {
    int idx = blockIdx.x * blockDim.x + threadIdx.x;
    int node = idx >> 5, c = idx & 31;
    if (node >= n) return;
    float4 v  = reinterpret_cast<float4*>(O + (size_t)node * 128)[c];
    float4 bb = reinterpret_cast<const float4*>(b)[c];
    v.x = fmaxf(v.x + bb.x, 0.f);
    v.y = fmaxf(v.y + bb.y, 0.f);
    v.z = fmaxf(v.z + bb.z, 0.f);
    v.w = fmaxf(v.w + bb.w, 0.f);
    reinterpret_cast<float4*>(O + (size_t)node * 128)[c] = v;
}

// ---------------- launch ----------------
extern "C" void kernel_launch(void* const* d_in, const int* in_sizes, int n_in,
                              void* d_out, int out_size, void* d_ws, size_t ws_size,
                              hipStream_t stream) {
    (void)in_sizes; (void)n_in; (void)out_size; (void)ws_size;
    const float* x   = (const float*)d_in[0];
    const int*   ei  = (const int*)d_in[1];
    const float* W1  = (const float*)d_in[2];
    const float* b1  = (const float*)d_in[3];
    const float* W2  = (const float*)d_in[4];
    const float* b2  = (const float*)d_in[5];
    const float* W3  = (const float*)d_in[6];
    const float* b3  = (const float*)d_in[7];
    const float* Wfc = (const float*)d_in[8];
    const float* bfc = (const float*)d_in[9];
    float* out = (float*)d_out;
    float* ws  = (float*)d_ws;

    const int n = NODES, e = EDGES;
    const int* src = ei;          // edge_index[0]
    const int* dst = ei + e;      // edge_index[1]

    float* dinv = ws;                       // N floats (deg -> dinv in place)
    float* buf0 = ws + 50048;               // N x 128
    float* buf1 = buf0 + (size_t)n * 128;   // N x 128

    const int nodeBlocks = (n + 255) / 256;
    const int vecBlocks  = (n * 32 + 255) / 256;   // node x float4-chunk grids
    const int edgeBlocks = (e + 255) / 256;
    const int scatBlocks = (e * 32 + 255) / 256;   // wave per edge
    const int gemmBlocks = (n + 127) / 128;

    // symmetric normalization: dinv = rsqrt(1 + indegree)
    k_deg_init <<<nodeBlocks, 256, 0, stream>>>(dinv, n);
    k_deg_count<<<edgeBlocks, 256, 0, stream>>>(dst, dinv, e);
    k_deg_rsqrt<<<nodeBlocks, 256, 0, stream>>>(dinv, n);

    const float* Hin = x;
    const float* Ws[3] = {W1, W2, W3};
    const float* bs[3] = {b1, b2, b3};
    for (int l = 0; l < 3; ++l) {
        k_gemm_bf16<128, false><<<gemmBlocks, 256, 0, stream>>>(Hin, Ws[l], nullptr, buf0, n);
        k_self_init   <<<vecBlocks,  256, 0, stream>>>(buf0, dinv, buf1, n);
        k_edge_scatter<<<scatBlocks, 256, 0, stream>>>(src, dst, dinv, buf0, buf1, e);
        k_bias_relu   <<<vecBlocks,  256, 0, stream>>>(buf1, bs[l], n);
        Hin = buf1;
    }
    // final FC: [N,128] @ [128,64] + bfc, no relu
    k_gemm_bf16<64, true><<<gemmBlocks, 256, 0, stream>>>(buf1, Wfc, bfc, out, n);
}
